// LiteHGATEncoder_10359461118487
// MI455X (gfx1250) — compile-verified, matrix operand order checked by reading, not run on hardware
//
#include <hip/hip_runtime.h>
#include <hip/hip_bf16.h>

// ---------------------------------------------------------------------------
// LiteHGATEncoder for MI455X (gfx1250, wave32).
// Heavy node GEMMs use v_wmma_f32_16x16x32_f16 (f16 operands, f32 accum).
// B weights are pre-packed into per-lane WMMA fragment order so both A and B
// fragments load as global_load_b128 with compile-time immediate offsets from
// a single per-wave base pointer. Each wave computes a 16x64 (or 16x32) strip
// sharing one A fragment across CT consecutive WMMAs.
// Edge-conditioned attention (segment softmax) via VALU + float atomics, with
// e = eattr @ We recomputed on the fly (edge_dim is only 2-3).
// ---------------------------------------------------------------------------

#define N_ORDER 40000
#define N_TRUCK 4000
#define N_DRONE 4000

typedef __attribute__((ext_vector_type(16))) _Float16     v16h;
typedef __attribute__((ext_vector_type(8)))  float        v8f;
typedef __attribute__((ext_vector_type(4)))  unsigned int u32x4;
typedef __attribute__((ext_vector_type(2)))  unsigned int u32x2;

// 16-bit WMMA operand layout: half j of lane -> K = (j%8) + 16*(j/8) + 8*(lane/16)
__device__ inline int wmma_k_of(int lane, int j) {
  return (j & 7) + ((j >> 3) << 4) + ((lane >> 4) << 3);
}

// ---------------------------------------------------------------------------
// Pack + convert B[128 x M] (f32, row-major) into WMMA fragment order (f16):
//   Bp[((ct*4 + ks)*32 + lane)*16 + j] = B[(ks*32 + K(lane,j))*M + ct*16 + (lane&15)]
// so each lane's 16 halves for a (colTile, kStep) fragment are contiguous (32B).
// ---------------------------------------------------------------------------
__global__ void pack_b_f16(const float* __restrict__ B, _Float16* __restrict__ Bp, int M) {
  int idx = blockIdx.x * blockDim.x + threadIdx.x;   // over 128*M elements
  if (idx >= 128 * M) return;
  const int j    = idx & 15;
  const int lane = (idx >> 4) & 31;
  const int frag = idx >> 9;         // ct*4 + ks
  const int ks   = frag & 3;
  const int ct   = frag >> 2;
  const int kk   = ks * 32 + wmma_k_of(lane, j);
  const int col  = ct * 16 + (lane & 15);
  Bp[idx] = (_Float16)B[(unsigned)(kk * M + col)];
}

// ---------------------------------------------------------------------------
// WMMA GEMM: Dst[N x M] = A[N x 128] (f16, row-major) @ B (packed f16) + bias
// grid.x = N/16 row tiles; wave w owns col tiles [w*CT, w*CT+CT).
// Launch with M/(16*CT) waves per block. M is a template constant so all
// strides are immediate offsets.
// ---------------------------------------------------------------------------
template <typename OutT, bool ACC, int CT, int M>
__global__ __launch_bounds__(256) void gemm_k128(
    const _Float16* __restrict__ A, const _Float16* __restrict__ Bp,
    const float* __restrict__ bias, OutT* __restrict__ Dst) {
  const int lane  = threadIdx.x & 31;
  const int wave  = threadIdx.x >> 5;
  const int rlane = lane & 15;     // A row / D col within tile
  const int hi    = lane >> 4;     // which K-half this lane holds
  const int rowT  = blockIdx.x;
  const int colT0 = wave * CT;

  // per-wave base pointers; everything below uses immediate offsets
  const _Float16* ap = A  + (unsigned)((rowT * 16 + rlane) * 128 + 8 * hi);
  const _Float16* bp = Bp + (unsigned)(colT0 * 2048 + lane * 16);

  union Frag { v16h h; u32x4 u[2]; };
  v8f acc[CT];
#pragma unroll
  for (int t = 0; t < CT; ++t) acc[t] = (v8f){};

#pragma unroll
  for (int ks = 0; ks < 4; ++ks) {       // K step: 32 per step
    Frag af;
    af.u[0] = *(const u32x4*)(ap + ks * 32);        // j=0..7
    af.u[1] = *(const u32x4*)(ap + ks * 32 + 16);   // j=8..15
#pragma unroll
    for (int t = 0; t < CT; ++t) {
      Frag bf;
      bf.u[0] = *(const u32x4*)(bp + t * 2048 + ks * 512);
      bf.u[1] = *(const u32x4*)(bp + t * 2048 + ks * 512 + 8);
      acc[t] = __builtin_amdgcn_wmma_f32_16x16x32_f16(
          /*neg_a=*/false, af.h, /*neg_b=*/false, bf.h,
          /*c_mod=*/(short)0, acc[t], /*reuse_a=*/false, /*reuse_b=*/false);
    }
  }

  // D layout: vgpr r -> row r (lanes 0-15) / row r+8 (lanes 16-31); col = lane&15
  OutT* drow = Dst + (unsigned)((rowT * 16 + 8 * hi) * M + colT0 * 16 + rlane);
#pragma unroll
  for (int t = 0; t < CT; ++t) {
    const float bv = bias[(unsigned)((colT0 + t) * 16 + rlane)];
#pragma unroll
    for (int r = 0; r < 8; ++r) {
      const float val = acc[t][r] + bv;
      if constexpr (ACC) {
        ((float*)drow)[r * M + t * 16] += val;
      } else {
        drow[r * M + t * 16] = (OutT)val;
      }
    }
  }
}

// ---------------------------------------------------------------------------
// small helper kernels
// ---------------------------------------------------------------------------
__global__ void zero_f32(float* __restrict__ p, int n) {
  int i = blockIdx.x * blockDim.x + threadIdx.x;
  if (i < n) p[i] = 0.f;
}

__global__ void init_mden(float* __restrict__ m, float* __restrict__ den, int n) {
  int i = blockIdx.x * blockDim.x + threadIdx.x;
  if (i < n) { m[i] = -__builtin_inff(); den[i] = 0.f; }
}

// proj: h32/h16 [N x 128] = x[N x Din] @ W[Din x 128] + b   (Din = 4 or 8)
__global__ void proj_kernel(const float* __restrict__ x, const float* __restrict__ W,
                            const float* __restrict__ b, float* __restrict__ h32,
                            _Float16* __restrict__ h16, int N, int Din) {
  int idx = blockIdx.x * blockDim.x + threadIdx.x;
  if (idx >= N * 128) return;
  int n = idx >> 7, c = idx & 127;
  float s = b[c];
  for (int d = 0; d < Din; ++d) s += x[n * Din + d] * W[d * 128 + c];
  h32[idx] = s;
  h16[idx] = (_Float16)s;
}

__device__ inline void atomicMaxF(float* addr, float val) {
  // orderable-int trick; init value is -inf (0xFF800000) so both paths compose
  if (val >= 0.f) atomicMax((int*)addr, __float_as_int(val));
  else            atomicMin((unsigned int*)addr, __float_as_uint(val));
}

// one wave per edge: score[e,h] = <q[dst,h,:], k[src,h,:]+e[h,:]> / sqrt(128)
// lane owns 4 contiguous channels -> q/k fragments load as 8-byte vectors.
__global__ __launch_bounds__(256) void edge_score(
    const _Float16* __restrict__ q, const _Float16* __restrict__ k,
    const int* __restrict__ src, const int* __restrict__ dst,
    const float* __restrict__ eattr, const float* __restrict__ We,
    float* __restrict__ scores, float* __restrict__ mbuf, int E, int ED) {
  const int e = blockIdx.x * 8 + (threadIdx.x >> 5);
  if (e >= E) return;
  const int lane = threadIdx.x & 31;
  const int c0 = lane * 4;
  const int sn = src[e], dn = dst[e];
  const _Float16* qp = q + (unsigned)(dn * 512) + c0;
  const _Float16* kp = k + (unsigned)(sn * 512) + c0;
  float ea[3];
  for (int d = 0; d < 3; ++d) ea[d] = (d < ED) ? eattr[(unsigned)(e * ED + d)] : 0.f;
  union H4 { u32x2 u; _Float16 h[4]; };
#pragma unroll
  for (int h = 0; h < 4; ++h) {
    H4 qv, kv;
    qv.u = *(const u32x2*)(qp + h * 128);
    kv.u = *(const u32x2*)(kp + h * 128);
    float s = 0.f;
#pragma unroll
    for (int i = 0; i < 4; ++i) {
      float ev = 0.f;
      for (int d = 0; d < ED; ++d) ev += ea[d] * We[(unsigned)(d * 512 + h * 128 + c0 + i)];
      s += (float)qv.h[i] * ((float)kv.h[i] + ev);
    }
#pragma unroll
    for (int off = 16; off > 0; off >>= 1) s += __shfl_xor(s, off, 32);
    if (lane == 0) {
      s *= 0.08838834764831845f;  // 1/sqrt(128)
      scores[(unsigned)(e * 4 + h)] = s;
      atomicMaxF(&mbuf[(unsigned)(dn * 4 + h)], s);
    }
  }
}

__global__ void edge_exp(float* __restrict__ scores, const float* __restrict__ mbuf,
                         float* __restrict__ den, const int* __restrict__ dst, int E) {
  int idx = blockIdx.x * blockDim.x + threadIdx.x;
  if (idx >= E * 4) return;
  int e = idx >> 2, h = idx & 3;
  int dn = dst[e];
  float a = __expf(scores[idx] - mbuf[(unsigned)(dn * 4 + h)]);
  scores[idx] = a;
  atomicAdd(&den[(unsigned)(dn * 4 + h)], a);
}

// one thread per (edge, channel-pair): acc[dst,c] += mean_h (v[src,h,c]+e[h,c])*a/denom
__global__ void edge_scatter(const _Float16* __restrict__ v,
                             const int* __restrict__ src, const int* __restrict__ dst,
                             const float* __restrict__ eattr, const float* __restrict__ We,
                             const float* __restrict__ scores, const float* __restrict__ den,
                             float* __restrict__ acc, int E, int ED) {
  int idx = blockIdx.x * blockDim.x + threadIdx.x;
  if (idx >= E * 64) return;
  int e = idx >> 6, c = (idx & 63) * 2;
  int sn = src[e], dn = dst[e];
  union H2 { unsigned u; _Float16 h[2]; };
  float s0 = 0.f, s1 = 0.f;
#pragma unroll
  for (int h = 0; h < 4; ++h) {
    float ev0 = 0.f, ev1 = 0.f;
    for (int d = 0; d < ED; ++d) {
      const float a = eattr[(unsigned)(e * ED + d)];
      ev0 += a * We[(unsigned)(d * 512 + h * 128 + c)];
      ev1 += a * We[(unsigned)(d * 512 + h * 128 + c + 1)];
    }
    H2 vv; vv.u = *(const unsigned*)(v + (unsigned)(sn * 512 + h * 128 + c));
    const float w = scores[(unsigned)(e * 4 + h)] / den[(unsigned)(dn * 4 + h)];
    s0 += ((float)vv.h[0] + ev0) * w;
    s1 += ((float)vv.h[1] + ev1) * w;
  }
  atomicAdd(&acc[(unsigned)(dn * 128 + c)],     0.25f * s0);
  atomicAdd(&acc[(unsigned)(dn * 128 + c + 1)], 0.25f * s1);
}

__global__ void relu_add_h(const float* __restrict__ h, const float* __restrict__ acc,
                           _Float16* __restrict__ out, int n) {
  int i = blockIdx.x * blockDim.x + threadIdx.x;
  if (i < n) { float v = h[i] + acc[i]; out[i] = (_Float16)(v > 0.f ? v : 0.f); }
}

// ---------------------------------------------------------------------------
extern "C" void kernel_launch(void* const* d_in, const int* in_sizes, int n_in,
                              void* d_out, int out_size, void* d_ws, size_t ws_size,
                              hipStream_t stream) {
  (void)in_sizes; (void)n_in; (void)out_size; (void)ws_size;

  // ---- input index map (dict insertion order, params recursively) ----
  const float* x_order = (const float*)d_in[0];
  const float* x_truck = (const float*)d_in[1];
  const float* x_drone = (const float*)d_in[2];
  // relations: t2o, o2t, d2o, o2d, o2o
  const int relEdge[5][3]  = {{3,4,5},{6,7,8},{9,10,11},{12,13,14},{15,16,17}};
  const int relBase[5]     = {24, 33, 42, 51, 60};   // Wq bq Wk bk Wv bv We Wskip bskip
  const int relED[5]       = {2, 2, 2, 2, 3};
  const int relSrcT[5]     = {1, 0, 2, 0, 0};        // 0=order 1=truck 2=drone
  const int relDstT[5]     = {0, 1, 0, 2, 0};
  const int relE[5]        = {80000, 80000, 80000, 80000, 160000};
  const int projW[3] = {18, 20, 22}, projB[3] = {19, 21, 23}, projD[3] = {8, 4, 4};
  const int postW[3] = {69, 71, 73}, postB[3] = {70, 72, 74};
  const int Ns[3] = {N_ORDER, N_TRUCK, N_DRONE};
  const int NMAX = N_ORDER, EMAX = 160000;

  // ---- carve workspace ----
  char* wsb = (char*)d_ws;
  size_t off = 0;
  auto alloc = [&](size_t bytes) -> char* {
    char* p = wsb + off;
    off = (off + bytes + 255) & ~(size_t)255;
    return p;
  };
  float*     h32[3]; _Float16* h16[3]; float* accb[3];
  for (int t = 0; t < 3; ++t) h32[t]  = (float*)alloc((size_t)Ns[t] * 128 * 4);
  for (int t = 0; t < 3; ++t) h16[t]  = (_Float16*)alloc((size_t)Ns[t] * 128 * 2);
  for (int t = 0; t < 3; ++t) accb[t] = (float*)alloc((size_t)Ns[t] * 128 * 4);
  _Float16* wq_h[5]; _Float16* wk_h[5]; _Float16* wv_h[5]; _Float16* wsk_h[5];
  for (int r = 0; r < 5; ++r) {
    wq_h[r]  = (_Float16*)alloc(128 * 512 * 2);
    wk_h[r]  = (_Float16*)alloc(128 * 512 * 2);
    wv_h[r]  = (_Float16*)alloc(128 * 512 * 2);
    wsk_h[r] = (_Float16*)alloc(128 * 128 * 2);
  }
  _Float16* wpo_h[3];
  for (int t = 0; t < 3; ++t) wpo_h[t] = (_Float16*)alloc(128 * 128 * 2);
  _Float16* qh = (_Float16*)alloc((size_t)NMAX * 512 * 2);
  _Float16* kh = (_Float16*)alloc((size_t)NMAX * 512 * 2);
  _Float16* vh = (_Float16*)alloc((size_t)NMAX * 512 * 2);
  float* scores = (float*)alloc((size_t)EMAX * 4 * 4);
  float* mbuf   = (float*)alloc((size_t)NMAX * 4 * 4);
  float* den    = (float*)alloc((size_t)NMAX * 4 * 4);
  _Float16* th  = (_Float16*)alloc((size_t)NMAX * 128 * 2);

  auto pack = [&](const void* s, _Float16* d, int M) {
    int n = 128 * M;
    pack_b_f16<<<(n + 255) / 256, 256, 0, stream>>>((const float*)s, d, M);
  };

  // ---- 0: convert + pack weights into WMMA fragment order (f16) ----
  for (int r = 0; r < 5; ++r) {
    pack(d_in[relBase[r] + 0], wq_h[r], 512);
    pack(d_in[relBase[r] + 2], wk_h[r], 512);
    pack(d_in[relBase[r] + 4], wv_h[r], 512);
    pack(d_in[relBase[r] + 7], wsk_h[r], 128);
  }
  for (int t = 0; t < 3; ++t) pack(d_in[postW[t]], wpo_h[t], 128);

  // ---- 1: input projections (tiny K, scalar) ----
  const float* xs[3] = {x_order, x_truck, x_drone};
  for (int t = 0; t < 3; ++t) {
    int n = Ns[t] * 128;
    proj_kernel<<<(n + 255) / 256, 256, 0, stream>>>(
        xs[t], (const float*)d_in[projW[t]], (const float*)d_in[projB[t]],
        h32[t], h16[t], Ns[t], projD[t]);
    zero_f32<<<(n + 255) / 256, 256, 0, stream>>>(accb[t], n);
  }

  // ---- 2: per-relation TransformerConv ----
  for (int r = 0; r < 5; ++r) {
    const int sT = relSrcT[r], dT = relDstT[r];
    const int Nsrc = Ns[sT], Ndst = Ns[dT], E = relE[r], ED = relED[r];
    const int* esrc = (const int*)d_in[relEdge[r][0]];
    const int* edst = (const int*)d_in[relEdge[r][1]];
    const float* eat = (const float*)d_in[relEdge[r][2]];
    const float* We  = (const float*)d_in[relBase[r] + 6];

    // Q/K/V: M=512 -> 8 waves x 4 col tiles per block
    gemm_k128<_Float16, false, 4, 512><<<Ndst / 16, 256, 0, stream>>>(
        h16[dT], wq_h[r], (const float*)d_in[relBase[r] + 1], qh);
    gemm_k128<_Float16, false, 4, 512><<<Nsrc / 16, 256, 0, stream>>>(
        h16[sT], wk_h[r], (const float*)d_in[relBase[r] + 3], kh);
    gemm_k128<_Float16, false, 4, 512><<<Nsrc / 16, 256, 0, stream>>>(
        h16[sT], wv_h[r], (const float*)d_in[relBase[r] + 5], vh);

    init_mden<<<(Ndst * 4 + 255) / 256, 256, 0, stream>>>(mbuf, den, Ndst * 4);
    edge_score<<<(E + 7) / 8, 256, 0, stream>>>(qh, kh, esrc, edst, eat, We,
                                                scores, mbuf, E, ED);
    edge_exp<<<(E * 4 + 255) / 256, 256, 0, stream>>>(scores, mbuf, den, edst, E);
    edge_scatter<<<(E * 64 + 255) / 256, 256, 0, stream>>>(
        vh, esrc, edst, eat, We, scores, den, accb[dT], E, ED);

    // root skip: acc += h_dst @ Wskip + bskip  (M=128 -> 4 waves x 2 col tiles)
    gemm_k128<float, true, 2, 128><<<Ndst / 16, 128, 0, stream>>>(
        h16[dT], wsk_h[r], (const float*)d_in[relBase[r] + 8], accb[dT]);
  }

  // ---- 3: z = relu(h + acc) @ Wpost + bpost ----
  float* outp = (float*)d_out;
  size_t oOff = 0;
  for (int t = 0; t < 3; ++t) {
    int n = Ns[t] * 128;
    relu_add_h<<<(n + 255) / 256, 256, 0, stream>>>(h32[t], accb[t], th, n);
    gemm_k128<float, false, 2, 128><<<Ns[t] / 16, 128, 0, stream>>>(
        th, wpo_h[t], (const float*)d_in[postB[t]], outp + oOff);
    oOff += (size_t)n;
  }
}
